// Model_6459630813787
// MI455X (gfx1250) — matrix operand order repaired and checked by reading, not compiled, not run on hardware
//
#include <hip/hip_runtime.h>
#include <hip/hip_bf16.h>

typedef __attribute__((ext_vector_type(16))) _Float16 v16h;
typedef __attribute__((ext_vector_type(8)))  _Float16 v8h;
typedef __attribute__((ext_vector_type(8)))  float    v8f;

#define NNODES 50000
#define DIN    128
#define DH     128
#define DOUT   64
#define NEDGE  640000
#define NEP    200000

// ---------------------------------------------------------------- utilities
__global__ __launch_bounds__(256) void k_zero_f32(float* p, int n) {
  int i = blockIdx.x * 256 + threadIdx.x;
  if (i < n) p[i] = 0.0f;
}

__global__ __launch_bounds__(256) void k_zero_u32(unsigned* p, int n) {
  int i = blockIdx.x * 256 + threadIdx.x;
  if (i < n) p[i] = 0u;
}

// ---------------------------------------------------------------- degrees
__global__ __launch_bounds__(256) void k_deg(const int* __restrict__ src,
                                             const int* __restrict__ dst,
                                             unsigned* cntOut, unsigned* cntIn, int E) {
  int e = blockIdx.x * 256 + threadIdx.x;
  if (e < E) {
    atomicAdd(&cntOut[src[e]], 1u);
    atomicAdd(&cntIn[dst[e]], 1u);
  }
}

__global__ __launch_bounds__(256) void k_deg_fin(const unsigned* __restrict__ cntOut,
                                                 const unsigned* __restrict__ cntIn,
                                                 float* dsout, float* dsin, int n) {
  int i = blockIdx.x * 256 + threadIdx.x;
  if (i < n) {
    dsout[i] = rsqrtf(fmaxf((float)cntOut[i], 1.0f));
    dsin[i]  = rsqrtf(fmaxf((float)cntIn[i],  1.0f));
  }
}

// ------------------------------------------------- pack W (K x N, row-major)
// into the WMMA B-fragment lane layout: wp[((nt*Ksteps+ks)*32 + lane)*16 + i]
// where for element (k, n): ks=k>>5, g=(k>>3)&1, i=(k&7)+8*((k>>4)&1), lane=(n&15)+16*g.
__global__ __launch_bounds__(256) void k_pack_w(const float* __restrict__ W,
                                                _Float16* __restrict__ wp, int K, int N) {
  int idx = blockIdx.x * 256 + threadIdx.x;
  if (idx >= K * N) return;
  int k = idx / N, n = idx - k * N;
  int ks = k >> 5, g = (k >> 3) & 1, i = (k & 7) + ((k >> 1) & 8);
  int nt = n >> 4, nl = n & 15;
  int Ksteps = K >> 5;
  int lane = nl + 16 * g;
  wp[(size_t)(((nt * Ksteps + ks) * 32) + lane) * 16 + i] = (_Float16)W[idx];
}

// ----------------------------------------- A operand: x * deg_out^-1/2 -> f16
__global__ __launch_bounds__(256) void k_scale_f16(const float* __restrict__ x,
                                                   const float* __restrict__ dsout,
                                                   _Float16* __restrict__ a16, int n, int F) {
  int i = blockIdx.x * 256 + threadIdx.x;
  if (i < n * F) {
    int row = i / F;
    a16[i] = (_Float16)(x[i] * dsout[row]);
  }
}

// ---------------------------------------------------------------- WMMA GEMM
// Y[M,N] (f32) = A[M,K] (f16, row-major) @ Wpacked[K,N] (fragment layout)
// one wave per 16x16 tile, Ksteps WMMA ops of 16x16x32.
__global__ __launch_bounds__(256) void k_wmma_gemm(const _Float16* __restrict__ A,
                                                   const _Float16* __restrict__ Wp,
                                                   float* __restrict__ Y,
                                                   int Mtiles, int Ntiles, int Ksteps, int N) {
  int wid = blockIdx.x * (256 >> 5) + (threadIdx.x >> 5);
  if (wid >= Mtiles * Ntiles) return;     // uniform per wave: EXEC stays all-ones
  int lane = threadIdx.x & 31;
  int mt = wid / Ntiles, nt = wid - mt * Ntiles;
  int ml = lane & 15, g = lane >> 4;

  int K = Ksteps * 32;
  const _Float16* arow  = A  + (size_t)(mt * 16 + ml) * K + g * 8;
  const _Float16* bbase = Wp + ((size_t)(nt * Ksteps) * 32 + lane) * 16;

  v8f c = {};
#pragma unroll
  for (int ks = 0; ks < 4; ++ks) {
    if (ks >= Ksteps) break;
    const _Float16* ap = arow + ks * 32;
    v8h lo = *(const v8h*)(ap);          // k = ks*32 + g*8 + [0..7]
    v8h hi = *(const v8h*)(ap + 16);     // k = ks*32 + 16 + g*8 + [0..7]
    v16h a;
#pragma unroll
    for (int i = 0; i < 8; ++i) { a[i] = lo[i]; a[i + 8] = hi[i]; }
    v16h b = *(const v16h*)(bbase + (size_t)ks * 32 * 16);
    c = __builtin_amdgcn_wmma_f32_16x16x32_f16(false, a, false, b, (short)0, c,
                                               false, false);
  }

  float* yout = Y + (size_t)mt * 16 * N + nt * 16 + ml;
#pragma unroll
  for (int r = 0; r < 8; ++r)
    yout[(size_t)(r + 8 * g) * N] = c[r];
}

// -------------------------------------------------- scatter: AGG[dst] += Y[src]
template <int F>
__global__ __launch_bounds__(256) void k_scatter(const float* __restrict__ Y,
                                                 const int* __restrict__ src,
                                                 const int* __restrict__ dst,
                                                 float* agg, int E) {
  int e = blockIdx.x * (256 >> 5) + (threadIdx.x >> 5);
  if (e >= E) return;
  int lane = threadIdx.x & 31;
  int s = src[e], d = dst[e];
  const float* yr = Y + (size_t)s * F;
  float* ar = agg + (size_t)d * F;
  __builtin_prefetch(ar, 1, 1);
#pragma unroll
  for (int j = 0; j < F / 32; ++j) {
    float v = yr[lane + j * 32];
    unsafeAtomicAdd(&ar[lane + j * 32], v);   // global_atomic_add_f32
  }
}

// --------------- finalize: h = relu(agg*dsin + b); emit next-layer f16 A or f32 h
__global__ __launch_bounds__(256) void k_finalize(const float* __restrict__ agg,
                                                  const float* __restrict__ dsin,
                                                  const float* __restrict__ bias,
                                                  const float* __restrict__ dsout,
                                                  _Float16* a16out, float* f32out,
                                                  int n, int F) {
  int i = blockIdx.x * 256 + threadIdx.x;
  if (i >= n * F) return;
  int row = i / F, f = i - row * F;
  float h = fmaxf(agg[i] * dsin[row] + bias[f], 0.0f);
  if (a16out) a16out[i] = (_Float16)(h * dsout[row]);
  if (f32out) f32out[i] = h;
}

// ------------------------------------------ edge dot: out[e] = <H[s], H[d]>
__global__ __launch_bounds__(256) void k_edge_dot(const float* __restrict__ H,
                                                  const int* __restrict__ es,
                                                  const int* __restrict__ ed,
                                                  float* out, int EP) {
  int e = blockIdx.x * (256 >> 5) + (threadIdx.x >> 5);
  if (e >= EP) return;
  int lane = threadIdx.x & 31;
  const float2* a = (const float2*)(H + (size_t)es[e] * DOUT);
  const float2* b = (const float2*)(H + (size_t)ed[e] * DOUT);
  float2 av = a[lane], bv = b[lane];
  float p = av.x * bv.x + av.y * bv.y;
#pragma unroll
  for (int off = 16; off > 0; off >>= 1) p += __shfl_xor(p, off, 32);
  if (lane == 0) out[e] = p;
}

// ---------------------------------------------------------------- launcher
extern "C" void kernel_launch(void* const* d_in, const int* in_sizes, int n_in,
                              void* d_out, int out_size, void* d_ws, size_t ws_size,
                              hipStream_t stream) {
  const float* x       = (const float*)d_in[0];
  const int*   src     = (const int*)d_in[1];
  const int*   dst     = (const int*)d_in[2];
  const int*   pos_src = (const int*)d_in[3];
  const int*   pos_dst = (const int*)d_in[4];
  const int*   neg_src = (const int*)d_in[5];
  const int*   neg_dst = (const int*)d_in[6];
  const float* W1      = (const float*)d_in[7];
  const float* b1      = (const float*)d_in[8];
  const float* W2      = (const float*)d_in[9];
  const float* b2      = (const float*)d_in[10];
  float* out = (float*)d_out;

  // ---- workspace carve-out (256B aligned chunks)
  char* base = (char*)d_ws;
  size_t off = 0;
  auto carve = [&](size_t bytes) -> void* {
    void* p = base + off;
    off = (off + bytes + 255) & ~(size_t)255;
    return p;
  };
  unsigned* cntOut = (unsigned*)carve(NNODES * 4);
  unsigned* cntIn  = (unsigned*)carve(NNODES * 4);
  float*    dsout  = (float*)carve(NNODES * 4);
  float*    dsin   = (float*)carve(NNODES * 4);
  _Float16* Wp1    = (_Float16*)carve((size_t)DIN * DH * 2);
  _Float16* Wp2    = (_Float16*)carve((size_t)DH * DOUT * 2);
  _Float16* A16    = (_Float16*)carve((size_t)NNODES * DIN * 2);
  float*    Y      = (float*)carve((size_t)NNODES * DH * 4);
  float*    AGG    = (float*)carve((size_t)NNODES * DH * 4);
  float*    H2     = (float*)carve((size_t)NNODES * DOUT * 4);
  (void)ws_size; (void)n_in; (void)in_sizes; (void)out_size;

  const int B = 256;
  auto blocks  = [](long n) { return (int)((n + 255) / 256); };
  auto wblocks = [](long waves) { return (int)((waves + 7) / 8); };

  // ---- degrees
  k_zero_u32<<<blocks(NNODES), B, 0, stream>>>(cntOut, NNODES);
  k_zero_u32<<<blocks(NNODES), B, 0, stream>>>(cntIn, NNODES);
  k_deg<<<blocks(NEDGE), B, 0, stream>>>(src, dst, cntOut, cntIn, NEDGE);
  k_deg_fin<<<blocks(NNODES), B, 0, stream>>>(cntOut, cntIn, dsout, dsin, NNODES);

  // ---- weight packing (fragment layout)
  k_pack_w<<<blocks((long)DIN * DH), B, 0, stream>>>(W1, Wp1, DIN, DH);
  k_pack_w<<<blocks((long)DH * DOUT), B, 0, stream>>>(W2, Wp2, DH, DOUT);

  // ---- layer 1: A = (x * dsout) f16 ; Y = A @ W1 (WMMA) ; AGG = segsum
  k_scale_f16<<<blocks((long)NNODES * DIN), B, 0, stream>>>(x, dsout, A16, NNODES, DIN);
  {
    int Mt = NNODES / 16, Nt = DH / 16, Ks = DIN / 32;
    k_wmma_gemm<<<wblocks((long)Mt * Nt), B, 0, stream>>>(A16, Wp1, Y, Mt, Nt, Ks, DH);
  }
  k_zero_f32<<<blocks((long)NNODES * DH), B, 0, stream>>>(AGG, NNODES * DH);
  k_scatter<DH><<<wblocks(NEDGE), B, 0, stream>>>(Y, src, dst, AGG, NEDGE);
  // h1 = relu(AGG*dsin + b1); A16 <- (h1 * dsout) f16  (layer-2 operand, fused)
  k_finalize<<<blocks((long)NNODES * DH), B, 0, stream>>>(AGG, dsin, b1, dsout,
                                                          A16, nullptr, NNODES, DH);

  // ---- layer 2
  {
    int Mt = NNODES / 16, Nt = DOUT / 16, Ks = DH / 32;
    k_wmma_gemm<<<wblocks((long)Mt * Nt), B, 0, stream>>>(A16, Wp2, Y, Mt, Nt, Ks, DOUT);
  }
  k_zero_f32<<<blocks((long)NNODES * DOUT), B, 0, stream>>>(AGG, NNODES * DOUT);
  k_scatter<DOUT><<<wblocks(NEDGE), B, 0, stream>>>(Y, src, dst, AGG, NEDGE);
  k_finalize<<<blocks((long)NNODES * DOUT), B, 0, stream>>>(AGG, dsin, b2, nullptr,
                                                            nullptr, H2, NNODES, DOUT);

  // ---- edge scores: out[0:EP) = pos, out[EP:2EP) = neg
  k_edge_dot<<<wblocks(NEP), B, 0, stream>>>(H2, pos_src, pos_dst, out, NEP);
  k_edge_dot<<<wblocks(NEP), B, 0, stream>>>(H2, neg_src, neg_dst, out + NEP, NEP);
}
